// ParametersGPT2_43533788512397
// MI455X (gfx1250) — matrix operand order repaired and checked by reading, not compiled, not run on hardware
//
#include <hip/hip_runtime.h>
#include <hip/hip_bf16.h>
#include <math.h>

typedef __attribute__((ext_vector_type(16))) __bf16 v16bf;
typedef __attribute__((ext_vector_type(8)))  float  v8f;
typedef __attribute__((ext_vector_type(4)))  unsigned int v4u;
typedef __attribute__((ext_vector_type(4)))  int v4i;
typedef __attribute__((ext_vector_type(8)))  int v8i;

#define nB   2
#define nS   1024
#define nD   768
#define nH   12
#define nDH  64
#define nDFF 3072
#define nV   50257
#define nEOS 50256
#define nL   12
#define nM   (nB * nS)   /* 2048 token rows */

#if __has_builtin(__builtin_amdgcn_tensor_load_to_lds) && \
    __has_builtin(__builtin_amdgcn_s_wait_tensorcnt)
#define USE_TDM 1
#else
#define USE_TDM 0
#endif

static __device__ __forceinline__ float bf2f(unsigned short u) {
  return __uint_as_float(((unsigned)u) << 16);
}
static __device__ __forceinline__ unsigned short f2bf(float f) {
  unsigned u = __float_as_uint(f);
  if ((u & 0x7fffffffu) > 0x7f800000u) return (unsigned short)0x7fc0u;  // NaN
  return (unsigned short)((u + (((u >> 16) & 1u) + 0x7fffu)) >> 16);     // RNE
}

union AF { v16bf v; unsigned short u[16]; };

static __device__ __forceinline__ v8f wmma_bf16(v16bf a, v16bf b, v8f c) {
  // D = A(16x32) * B(32x16) + C(16x16 f32)
  return __builtin_amdgcn_wmma_f32_16x16x32_bf16(false, a, false, b, (short)0, c,
                                                 false, false);
}

#if USE_TDM
// Issue a TDM 2D tile load (bf16 elements) into padded LDS.
// Row = tile_d0 elements (64B when tile_d0==32); LDS rows padded +16B
// (pad_interval code 3 => every 16 DWORDs, pad_amount code 3 => 4 DWORDs),
// matching a [rows][40]-ushort LDS tile.
static __device__ __forceinline__ void tdm_load_2d(
    unsigned lds_addr, unsigned long long gaddr, unsigned tensor_d0,
    unsigned tensor_d1, unsigned tile_d0, unsigned tile_d1,
    unsigned long long stride0_elems) {
  v4u g0;
  g0.x = 1u;                                        // count=1 valid user D#
  g0.y = lds_addr;                                  // LDS byte address
  g0.z = (unsigned)(gaddr & 0xffffffffu);           // global_addr[31:0]
  g0.w = (unsigned)((gaddr >> 32) & 0x01ffffffu) |  // global_addr[56:32]
         (2u << 30);                                // type=2 ("image")
  v8i g1;
  g1[0] = (int)((1u << 16) |        // data_size = 2 bytes
                (1u << 20) |        // pad_enable
                (3u << 22) |        // pad_interval: 16 DWORDs (64B)
                (3u << 25));        // pad_amount: 4 DWORDs (16B)
  g1[1] = (int)((tensor_d0 & 0xffffu) << 16);
  g1[2] = (int)(((tensor_d0 >> 16) & 0xffffu) | ((tensor_d1 & 0xffffu) << 16));
  g1[3] = (int)(((tensor_d1 >> 16) & 0xffffu) | ((tile_d0 & 0xffffu) << 16));
  g1[4] = (int)(tile_d1 & 0xffffu);                 // tile_dim1 (tile_dim2=0)
  g1[5] = (int)(stride0_elems & 0xffffffffu);       // tensor_dim0_stride[31:0]
  g1[6] = (int)((stride0_elems >> 32) & 0xffffu);   // stride[47:32], dim1_stride lo=0
  g1[7] = 0;
  v4i z4 = {};
#if __clang_major__ >= 23
  v8i z8 = {};
  __builtin_amdgcn_tensor_load_to_lds(g0, g1, z4, z4, z8, 0);
#else
  __builtin_amdgcn_tensor_load_to_lds(g0, g1, z4, z4, 0);
#endif
}
#endif  // USE_TDM

// ---------------- segment scan: last-EOS cummax + EOS cumsum ----------------
__global__ void scan_kernel(const int* __restrict__ ids, int* __restrict__ rel,
                            int* __restrict__ seg) {
  if (threadIdx.x != 0) return;
  int b = blockIdx.x;
  int last = 0, sc = 0;
  for (int s = 0; s < nS; ++s) {
    int id = ids[b * nS + s];
    if (id == nEOS) { last = s; ++sc; }
    rel[b * nS + s] = s - last;
    seg[b * nS + s] = sc;
  }
}

// ---------------- embedding: tok*sqrt(D) + pos[rel] ----------------
__global__ void embed_kernel(const int* __restrict__ ids, const int* __restrict__ rel,
                             const unsigned short* __restrict__ tok,
                             const unsigned short* __restrict__ pos,
                             unsigned short* __restrict__ x) {
  int bs = blockIdx.x;
  int id = ids[bs], rl = rel[bs];
  const float sq = 27.712812921102035f;  // sqrt(768)
  for (int d = threadIdx.x; d < nD; d += blockDim.x)
    x[(size_t)bs * nD + d] =
        f2bf(bf2f(tok[(size_t)id * nD + d]) * sq + bf2f(pos[(size_t)rl * nD + d]));
}

// ---------------- fp32 LayerNorm (biased variance) ----------------
__global__ __launch_bounds__(256) void layernorm_kernel(
    const unsigned short* __restrict__ x, const float* __restrict__ w,
    const float* __restrict__ bpar, unsigned short* __restrict__ out) {
  __shared__ float red[256];
  int t = threadIdx.x;
  size_t row = blockIdx.x;
  const unsigned short* xr = x + row * nD;
  float v0 = bf2f(xr[t]), v1 = bf2f(xr[t + 256]), v2 = bf2f(xr[t + 512]);
  red[t] = v0 + v1 + v2;
  __syncthreads();
  for (int o = 128; o > 0; o >>= 1) { if (t < o) red[t] += red[t + o]; __syncthreads(); }
  float mean = red[0] * (1.0f / nD);
  __syncthreads();
  float d0 = v0 - mean, d1 = v1 - mean, d2 = v2 - mean;
  red[t] = d0 * d0 + d1 * d1 + d2 * d2;
  __syncthreads();
  for (int o = 128; o > 0; o >>= 1) { if (t < o) red[t] += red[t + o]; __syncthreads(); }
  float rstd = rsqrtf(red[0] * (1.0f / nD) + 1e-5f);
  unsigned short* orow = out + row * nD;
  orow[t]       = f2bf(w[t]       * d0 * rstd + bpar[t]);
  orow[t + 256] = f2bf(w[t + 256] * d1 * rstd + bpar[t + 256]);
  orow[t + 512] = f2bf(w[t + 512] * d2 * rstd + bpar[t + 512]);
}

// ---------------- WMMA GEMM: C[M,N] = A[M,K] @ W[N,K]^T (+bias)(+GELU)(+resid) ----
// Block tile 128(M) x 64(N), BK=32; 8 waves in a 4x2 grid, each wave does 32x32.
// TDM path: double-buffered tensor_load_to_lds (wave 0 issues, TENSORcnt pipelined).
__global__ __launch_bounds__(256) void gemm_kernel(
    const unsigned short* __restrict__ A, const unsigned short* __restrict__ W,
    const unsigned short* __restrict__ bias, const unsigned short* __restrict__ resid,
    unsigned short* __restrict__ C, int M, int N, int Kd, int act) {
  __shared__ unsigned short As[2][128][40];  // 64B row + 16B pad
  __shared__ unsigned short Bs[2][64][40];
  const int tid = threadIdx.x;
  const int m0 = blockIdx.y * 128, n0 = blockIdx.x * 64;
  const int wid = tid >> 5, lane = tid & 31, half = lane >> 4, ln = lane & 15;
  const int wm = (wid & 3) * 32, wn = (wid >> 2) * 32;
  const int nk = Kd / 32;

#if USE_TDM
  const unsigned ldsA[2] = {(unsigned)(unsigned long long)&As[0][0][0],
                            (unsigned)(unsigned long long)&As[1][0][0]};
  const unsigned ldsB[2] = {(unsigned)(unsigned long long)&Bs[0][0][0],
                            (unsigned)(unsigned long long)&Bs[1][0][0]};
  const unsigned long long gA = (unsigned long long)A + (size_t)m0 * Kd * 2;
  const unsigned long long gB = (unsigned long long)W + (size_t)n0 * Kd * 2;
  const unsigned remA = (unsigned)(M - m0);
  const unsigned remB = (unsigned)(N - n0);   // < 64 at the ragged logits edge -> OOB rows read 0
  if (tid < 32) {  // prologue: tile 0 in flight
    tdm_load_2d(ldsA[0], gA, (unsigned)Kd, remA, 32u, 128u, (unsigned long long)Kd);
    tdm_load_2d(ldsB[0], gB, (unsigned)Kd, remB, 32u, 64u, (unsigned long long)Kd);
  }
#else
  const int arow = tid >> 1, ahalf = tid & 1;   // 2 threads / A row, 16 cols each
  const int brow = tid >> 2, bq = tid & 3;      // 4 threads / B row, 8 cols each
  int gmA = m0 + arow; if (gmA > M - 1) gmA = M - 1;
  int gnB = n0 + brow; if (gnB > N - 1) gnB = N - 1;
  const unsigned short* asrc = A + (size_t)gmA * Kd + ahalf * 16;
  const unsigned short* bsrc = W + (size_t)gnB * Kd + bq * 8;
#endif

  v8f acc[2][2] = {};
  for (int it = 0; it < nk; ++it) {
#if USE_TDM
    const int buf = it & 1;
    if (tid < 32) {
      if (it + 1 < nk) {  // issue next tile pair, then wait for current pair
        unsigned long long koff = (unsigned long long)(it + 1) * 64;  // 32 elems * 2B
        unsigned kremain = (unsigned)(Kd - (it + 1) * 32);
        tdm_load_2d(ldsA[buf ^ 1], gA + koff, kremain, remA, 32u, 128u,
                    (unsigned long long)Kd);
        tdm_load_2d(ldsB[buf ^ 1], gB + koff, kremain, remB, 32u, 64u,
                    (unsigned long long)Kd);
        __builtin_amdgcn_s_wait_tensorcnt(2);
      } else {
        __builtin_amdgcn_s_wait_tensorcnt(0);
      }
    }
    __syncthreads();
#else
    const int buf = 0;
    const int k0 = it * 32;
    {
      const uint4* ap4 = (const uint4*)(asrc + k0);
      uint4 a0 = ap4[0], a1 = ap4[1];
      uint4 b0 = *(const uint4*)(bsrc + k0);
      *(uint4*)&As[0][arow][ahalf * 16]     = a0;
      *(uint4*)&As[0][arow][ahalf * 16 + 8] = a1;
      *(uint4*)&Bs[0][brow][bq * 8]         = b0;
    }
    __syncthreads();
    if (k0 + 32 < Kd) {
      __builtin_prefetch(asrc + k0 + 32, 0, 1);
      __builtin_prefetch(bsrc + k0 + 32, 0, 1);
    }
#endif
    AF afr[2], bfr[2];
#pragma unroll
    for (int i = 0; i < 2; ++i) {
      int row = wm + i * 16 + ln;
#pragma unroll
      for (int j = 0; j < 16; ++j) {            // ISA A layout: K=16*(j/8)+8*half+(j%8)
        int kk = 16 * (j >> 3) + half * 8 + (j & 7);
        afr[i].u[j] = As[buf][row][kk];
      }
    }
#pragma unroll
    for (int t = 0; t < 2; ++t) {
      int col = wn + t * 16 + ln;
#pragma unroll
      for (int j = 0; j < 16; ++j)              // ISA B layout: K=16*(j/8)+8*half+(j%8)
        bfr[t].u[j] = Bs[buf][col][16 * (j >> 3) + half * 8 + (j & 7)];
    }
#pragma unroll
    for (int i = 0; i < 2; ++i)
#pragma unroll
      for (int t = 0; t < 2; ++t)
        acc[i][t] = wmma_bf16(afr[i].v, bfr[t].v, acc[i][t]);
    __syncthreads();
  }
#pragma unroll
  for (int i = 0; i < 2; ++i)
#pragma unroll
    for (int t = 0; t < 2; ++t)
#pragma unroll
      for (int r = 0; r < 8; ++r) {             // C layout: M=r+8*half, N=ln
        int gm = m0 + wm + i * 16 + r + half * 8;
        int gn = n0 + wn + t * 16 + ln;
        if (gn < N && gm < M) {
          float v = acc[i][t][r];
          if (bias) v += bf2f(bias[gn]);
          if (act == 1) v = 0.5f * v * (1.0f + erff(v * 0.70710678118654752f));
          if (resid) v += bf2f(resid[(size_t)gm * N + gn]);
          C[(size_t)gm * N + gn] = f2bf(v);
        }
      }
}

// ---------------- split interleaved QKV [B,S,(H,DH,3)] -> Q/K/V [B,H,S,DH] ------
__global__ void split_kernel(const unsigned short* __restrict__ qkv,
                             unsigned short* __restrict__ Q,
                             unsigned short* __restrict__ K,
                             unsigned short* __restrict__ V) {
  int bs = blockIdx.x;
  int b = bs / nS, s = bs % nS;
  const unsigned short* src = qkv + (size_t)bs * (3 * nD);
  for (int n = threadIdx.x; n < 3 * nD; n += blockDim.x) {
    int h = n / (nDH * 3);
    int rem = n % (nDH * 3);
    int dh = rem / 3, c = rem % 3;
    unsigned short* dst = (c == 0) ? Q : (c == 1) ? K : V;
    dst[(((size_t)b * nH + h) * nS + s) * nDH + dh] = src[n];
  }
}

// ---------------- flash attention, WMMA for QK^T and P@V ----------------
// 1 wave per 16-query tile; 4 waves/block; key tiles of 32 with online softmax.
__global__ __launch_bounds__(128) void attention_kernel(
    const unsigned short* __restrict__ Q, const unsigned short* __restrict__ Km,
    const unsigned short* __restrict__ Vm, const int* __restrict__ seg,
    unsigned short* __restrict__ ctx) {
  __shared__ unsigned short plds[4][16 * 32];   // per-wave P tile (bf16 16x32)
  const int wv = threadIdx.x >> 5, lane = threadIdx.x & 31;
  const int half = lane >> 4, ln = lane & 15;
  const int bh = blockIdx.y, b = bh / nH, h = bh % nH;
  const int q0 = (blockIdx.x * 4 + wv) * 16;
  const unsigned short* Qb = Q + (size_t)bh * nS * nDH;
  const unsigned short* Kb = Km + (size_t)bh * nS * nDH;
  const unsigned short* Vb = Vm + (size_t)bh * nS * nDH;
  const int* segb = seg + b * nS;

  AF aq[2];
  {
    int qrow = q0 + ln;
#pragma unroll
    for (int c = 0; c < 2; ++c)
#pragma unroll
      for (int j = 0; j < 16; ++j) {
        int kk = c * 32 + 16 * (j >> 3) + half * 8 + (j & 7);
        aq[c].u[j] = Qb[(size_t)qrow * nDH + kk];
      }
  }
  int segq[8];
#pragma unroll
  for (int r = 0; r < 8; ++r) segq[r] = segb[q0 + r + half * 8];

  v8f accv[4] = {};
  float mrow[8], lrow[8];
#pragma unroll
  for (int r = 0; r < 8; ++r) { mrow[r] = -1e30f; lrow[r] = 0.0f; }
  const float scale = 0.125f;  // 1/sqrt(64)

  for (int k0 = 0; k0 <= q0 + 15; k0 += 32) {
    v8f s0 = {}, s1 = {};
#pragma unroll
    for (int c = 0; c < 2; ++c) {
      AF bk0, bk1;
      int key0 = k0 + ln, key1 = k0 + 16 + ln;
      int ck0 = key0 < nS ? key0 : nS - 1;
      int ck1 = key1 < nS ? key1 : nS - 1;
#pragma unroll
      for (int j = 0; j < 16; ++j) {
        int dh = c * 32 + 16 * (j >> 3) + half * 8 + (j & 7);
        bk0.u[j] = Kb[(size_t)ck0 * nDH + dh];
        bk1.u[j] = Kb[(size_t)ck1 * nDH + dh];
      }
      s0 = wmma_bf16(aq[c].v, bk0.v, s0);
      s1 = wmma_bf16(aq[c].v, bk1.v, s1);
    }
    int key0 = k0 + ln, key1 = k0 + 16 + ln;
    int sk0 = segb[key0 < nS ? key0 : nS - 1];
    int sk1 = segb[key1 < nS ? key1 : nS - 1];
    float p0s[8], p1s[8];
#pragma unroll
    for (int r = 0; r < 8; ++r) {
      int qi = q0 + r + half * 8;
      float e0 = s0[r] * scale + ((sk0 == segq[r] && qi >= key0) ? 0.0f : -1e30f);
      float e1 = s1[r] * scale + ((sk1 == segq[r] && qi >= key1) ? 0.0f : -1e30f);
      float rm = fmaxf(e0, e1);
      rm = fmaxf(rm, __shfl_xor(rm, 1));
      rm = fmaxf(rm, __shfl_xor(rm, 2));
      rm = fmaxf(rm, __shfl_xor(rm, 4));
      rm = fmaxf(rm, __shfl_xor(rm, 8));        // reduce within 16-lane half
      float mnew = fmaxf(mrow[r], rm);
      float alpha = __expf(mrow[r] - mnew);
      float p0 = __expf(e0 - mnew), p1 = __expf(e1 - mnew);
      float rs = p0 + p1;
      rs += __shfl_xor(rs, 1);
      rs += __shfl_xor(rs, 2);
      rs += __shfl_xor(rs, 4);
      rs += __shfl_xor(rs, 8);
      lrow[r] = lrow[r] * alpha + rs;
      mrow[r] = mnew;
#pragma unroll
      for (int t = 0; t < 4; ++t) accv[t][r] *= alpha;
      p0s[r] = p0; p1s[r] = p1;
    }
    // C-layout -> A-layout re-stripe via per-wave LDS round-trip
    unsigned short* pl = plds[wv];
#pragma unroll
    for (int r = 0; r < 8; ++r) {
      int row = r + half * 8;
      pl[row * 32 + ln]      = f2bf(p0s[r]);
      pl[row * 32 + 16 + ln] = f2bf(p1s[r]);
    }
    AF ap;
#pragma unroll
    for (int j = 0; j < 16; ++j) {
      int kk = 16 * (j >> 3) + half * 8 + (j & 7);
      ap.u[j] = pl[ln * 32 + kk];
    }
#pragma unroll
    for (int t = 0; t < 4; ++t) {
      AF bv;
#pragma unroll
      for (int j = 0; j < 16; ++j) {
        int kk = 16 * (j >> 3) + half * 8 + (j & 7);
        int key = k0 + kk; if (key >= nS) key = nS - 1;
        bv.u[j] = Vb[(size_t)key * nDH + t * 16 + ln];
      }
      accv[t] = wmma_bf16(ap.v, bv.v, accv[t]);
    }
  }
#pragma unroll
  for (int t = 0; t < 4; ++t)
#pragma unroll
    for (int r = 0; r < 8; ++r) {
      int qi = q0 + r + half * 8;
      float o = accv[t][r] / lrow[r];
      ctx[((size_t)b * nS + qi) * nD + h * nDH + t * 16 + ln] = f2bf(o);
    }
}

// ---------------- host orchestration ----------------
extern "C" void kernel_launch(void* const* d_in, const int* in_sizes, int n_in,
                              void* d_out, int out_size, void* d_ws, size_t ws_size,
                              hipStream_t stream) {
  const int*            ids  = (const int*)d_in[0];
  const unsigned short* tok  = (const unsigned short*)d_in[1];
  const unsigned short* pos  = (const unsigned short*)d_in[2];
  const float*          ln1w = (const float*)d_in[3];
  const float*          ln1b = (const float*)d_in[4];
  const unsigned short* qkvw = (const unsigned short*)d_in[5];
  const unsigned short* qkvb = (const unsigned short*)d_in[6];
  const unsigned short* outw = (const unsigned short*)d_in[7];
  const unsigned short* outb = (const unsigned short*)d_in[8];
  const float*          ln2w = (const float*)d_in[9];
  const float*          ln2b = (const float*)d_in[10];
  const unsigned short* fc1w = (const unsigned short*)d_in[11];
  const unsigned short* fc1b = (const unsigned short*)d_in[12];
  const unsigned short* fc2w = (const unsigned short*)d_in[13];
  const unsigned short* fc2b = (const unsigned short*)d_in[14];

  char* base = (char*)d_ws;
  size_t off = 0;
  auto alloc = [&](size_t bytes) {
    void* p = base + off;
    off = (off + bytes + 255) & ~(size_t)255;
    return p;
  };
  const size_t BSD = (size_t)nM * nD * 2;
  unsigned short* x    = (unsigned short*)alloc(BSD);
  unsigned short* hbuf = (unsigned short*)alloc(BSD);
  unsigned short* qkv  = (unsigned short*)alloc((size_t)nM * 3 * nD * 2);
  unsigned short* Qb   = (unsigned short*)alloc(BSD);
  unsigned short* Kb   = (unsigned short*)alloc(BSD);
  unsigned short* Vb   = (unsigned short*)alloc(BSD);
  unsigned short* ctx  = (unsigned short*)alloc(BSD);
  unsigned short* ff   = (unsigned short*)alloc((size_t)nM * nDFF * 2);
  int* rel = (int*)alloc((size_t)nM * 4);
  int* seg = (int*)alloc((size_t)nM * 4);

  scan_kernel<<<nB, 32, 0, stream>>>(ids, rel, seg);
  embed_kernel<<<nM, 256, 0, stream>>>(ids, rel, tok, pos, x);

  auto gemm = [&](const unsigned short* A, const unsigned short* W,
                  const unsigned short* bias, const unsigned short* resid,
                  unsigned short* C, int N, int Kd, int act) {
    dim3 g((N + 63) / 64, nM / 128);
    gemm_kernel<<<g, 256, 0, stream>>>(A, W, bias, resid, C, nM, N, Kd, act);
  };

  for (int i = 0; i < nL; ++i) {
    layernorm_kernel<<<nM, 256, 0, stream>>>(x, ln1w + (size_t)i * nD,
                                             ln1b + (size_t)i * nD, hbuf);
    gemm(hbuf, qkvw + (size_t)i * 3 * nD * nD, qkvb + (size_t)i * 3 * nD,
         nullptr, qkv, 3 * nD, nD, 0);
    split_kernel<<<nM, 256, 0, stream>>>(qkv, Qb, Kb, Vb);
    attention_kernel<<<dim3(nS / 64, nB * nH), 128, 0, stream>>>(Qb, Kb, Vb, seg, ctx);
    gemm(ctx, outw + (size_t)i * nD * nD, outb + (size_t)i * nD, x, x, nD, nD, 0);
    layernorm_kernel<<<nM, 256, 0, stream>>>(x, ln2w + (size_t)i * nD,
                                             ln2b + (size_t)i * nD, hbuf);
    gemm(hbuf, fc1w + (size_t)i * nDFF * nD, fc1b + (size_t)i * nDFF,
         nullptr, ff, nDFF, nD, 1);
    gemm(ff, fc2w + (size_t)i * nD * nDFF, fc2b + (size_t)i * nD, x, x, nD, nDFF, 0);
  }
  // tied decoder: logits = x @ tok_emb^T  (N = 50257 with edge guards)
  gemm(x, tok, nullptr, nullptr, (unsigned short*)d_out, nV, nD, 0);

  (void)in_sizes; (void)n_in; (void)out_size; (void)ws_size;
}
